// EdgeTypeRGCN_27522150432768
// MI455X (gfx1250) — matrix operand order repaired and checked by reading, compiled, not run on hardware
//
#include <hip/hip_runtime.h>
#include <hip/hip_bf16.h>

// ---------------------------------------------------------------------------
// RGCN (basis decomposition) on gfx1250, bf16 WMMA pipeline.
//
//   1) build_wt:   Wt[r][o][i] = bf16( sum_b comp[r,b]*bases[b][i][o] ),
//                  Wt[R][o][i] = bf16( loop_weight[i][o] )      (transposed!)
//   2) to_bf16:    node_feats -> bf16 copy (25.6MB, L2-resident thereafter)
//   3) gemm_bf16:  C[m][o] = sum_i A[m][i] * Wt[w][o][i]  via
//                  v_wmma_f32_16x16x32_bf16 (LDS-staged A, f32 accumulate)
//        - self-loop pass writes h = C + bias
//        - per-relation pass writes T (51MB, stays in 192MB L2)
//   4) scatter_rel: edges of relation r: h[dst] += T[src] (global_atomic_add_f32,
//                  hits L2 since T and h are both L2-resident)
//   5) finalize:   LeakyReLU(0.1) + LayerNorm (wave32 shuffle reduction)
// ---------------------------------------------------------------------------

typedef __attribute__((ext_vector_type(16))) __bf16 bf16x16;
typedef __attribute__((ext_vector_type(8)))  float  f32x8;

#define FEAT 256
#define NEG_SLOPE 0.1f
#define LN_EPS 1e-5f

union FragBF { bf16x16 v; uint4 q[2]; };

__device__ __forceinline__ unsigned short f32_to_bf16_rne(float f) {
  unsigned int u = __float_as_uint(f);
  unsigned int r = u + 0x7FFFu + ((u >> 16) & 1u);
  return (unsigned short)(r >> 16);
}

// ---- 1) build transposed bf16 weights: Wt[(R+1)][FEAT(out)][FEAT(in)] ------
__global__ void build_wt_kernel(const float* __restrict__ bases,
                                const float* __restrict__ comp,
                                const float* __restrict__ loopW,
                                unsigned short* __restrict__ Wt,
                                int R, int Bn) {
  int id = blockIdx.x * blockDim.x + threadIdx.x;
  int total = (R + 1) * FEAT * FEAT;
  if (id >= total) return;
  int widx = id / (FEAT * FEAT);
  int rem  = id % (FEAT * FEAT);
  int o = rem / FEAT;
  int i = rem % FEAT;
  float w;
  if (widx < R) {
    w = 0.0f;
    for (int b = 0; b < Bn; ++b)
      w += comp[widx * Bn + b] * bases[((size_t)b * FEAT + i) * FEAT + o];
  } else {
    w = loopW[(size_t)i * FEAT + o];
  }
  Wt[(size_t)widx * FEAT * FEAT + (size_t)o * FEAT + i] = f32_to_bf16_rne(w);
}

// ---- 2) f32 -> bf16 copy ---------------------------------------------------
__global__ void to_bf16_kernel(const float* __restrict__ x,
                               unsigned short* __restrict__ y, int n) {
  int id = blockIdx.x * blockDim.x + threadIdx.x;
  if (id < n) y[id] = f32_to_bf16_rne(x[id]);
}

// ---- 3) GEMM: out[m][o] = sum_i A[m][i] * Wt[widx][o][i] -------------------
// Block: 256 threads = 8 waves; tile: 64 rows x 256 cols.
// wave w: row-block rb = w&3 (16 rows), col-half ch = w>>2 (128 cols = 8 subtiles).
__global__ void gemm_bf16_kernel(const unsigned short* __restrict__ nodeBf,
                                 const unsigned short* __restrict__ Wt,
                                 int widx,
                                 float* __restrict__ out,
                                 const float* __restrict__ bias,
                                 int addBias, int Nn) {
  __shared__ __align__(16) unsigned short smemA[64 * 32];  // 4KB A tile
  uint4* smq = reinterpret_cast<uint4*>(smemA);

  const int tid  = threadIdx.x;
  const int wave = tid >> 5;
  const int lane = tid & 31;
  const int l16  = lane & 15;
  const int kh   = lane >> 4;      // lane half (K-half selector)
  const int rb   = wave & 3;       // row block within 64-row tile
  const int ch   = wave >> 2;      // column half (0/1 -> 128 cols each)
  const int m0   = blockIdx.x * 64;

  const unsigned short* W0 = Wt + (size_t)widx * FEAT * FEAT;

  f32x8 acc[8];
  const f32x8 zero8 = {0.f, 0.f, 0.f, 0.f, 0.f, 0.f, 0.f, 0.f};
#pragma unroll
  for (int s = 0; s < 8; ++s) acc[s] = zero8;

  const int fillRow   = tid >> 2;   // 0..63
  const int fillChunk = tid & 3;    // 0..3 (16B chunks of a 64B row)

  for (int kk = 0; kk < FEAT / 32; ++kk) {
    const int k0 = kk * 32;

    // ---- stage A tile [64 x 32] bf16 into LDS (coalesced b128) ----
    __syncthreads();
    {
      const int g = m0 + fillRow;
      uint4 val = make_uint4(0u, 0u, 0u, 0u);
      if (g < Nn) {
        const uint4* src = reinterpret_cast<const uint4*>(
            nodeBf + (size_t)g * FEAT + k0 + fillChunk * 8);
        val = *src;
      }
      smq[tid] = val;
    }
    __syncthreads();

    // ---- A fragment: documented 16-bit A 16x32 layout ----
    // lane half kh: K in {kh*8 .. kh*8+7} and {kh*8+16 .. kh*8+23}
    FragBF afrag;
    {
      const int ml = rb * 16 + l16;          // local row
      afrag.q[0] = smq[ml * 4 + kh];         // K = kh*8 + 0..7
      afrag.q[1] = smq[ml * 4 + kh + 2];     // K = kh*8 + 16..23
    }

    // ---- 8 subtiles of 16 columns: B fragment is two 16B reads ---
#pragma unroll
    for (int s = 0; s < 8; ++s) {
      const int col = ch * 128 + s * 16 + l16;
      const uint4* bq = reinterpret_cast<const uint4*>(
          W0 + (size_t)col * FEAT + k0);   // K = kh*8+0..7 and kh*8+16..23
      FragBF bfrag;
      bfrag.q[0] = bq[kh];
      bfrag.q[1] = bq[kh + 2];
      acc[s] = __builtin_amdgcn_wmma_f32_16x16x32_bf16(
          /*neg_a=*/false, afrag.v, /*neg_b=*/false, bfrag.v,
          /*c_mod=*/(short)0, acc[s], /*reuse_a=*/false, /*reuse_b=*/false);
    }
  }

  // ---- store C: lane holds col = l16, rows kh*8 .. kh*8+7 of its 16x16 tile
  const int mrowBase = m0 + rb * 16 + kh * 8;
#pragma unroll
  for (int s = 0; s < 8; ++s) {
    const int col = ch * 128 + s * 16 + l16;
    const float bv = addBias ? bias[col] : 0.0f;
#pragma unroll
    for (int j = 0; j < 8; ++j) {
      const int row = mrowBase + j;
      if (row < Nn) out[(size_t)row * FEAT + col] = acc[s][j] + bv;
    }
  }
}

// ---- 4) scatter: for edges of relation `rel`: h[dst] += T[src] -------------
__global__ void scatter_rel_kernel(const float* __restrict__ T,
                                   float* __restrict__ h,
                                   const int* __restrict__ src,
                                   const int* __restrict__ dst,
                                   const int* __restrict__ etype,
                                   int E, int rel) {
  const int lane = threadIdx.x & 31;
  const int waveInBlock = threadIdx.x >> 5;
  const int wavesPerBlock = blockDim.x >> 5;
  const long long totalWaves = (long long)gridDim.x * wavesPerBlock;
  for (long long e = (long long)blockIdx.x * wavesPerBlock + waveInBlock;
       e < E; e += totalWaves) {
    if (etype[e] != rel) continue;
    const size_t sOff = (size_t)src[e] * FEAT;
    const size_t dOff = (size_t)dst[e] * FEAT;
#pragma unroll
    for (int i = 0; i < FEAT / 32; ++i) {
      const int c = lane + 32 * i;
      unsafeAtomicAdd(&h[dOff + c], T[sOff + c]);   // global_atomic_add_f32
    }
  }
}

// ---- 5) LeakyReLU + LayerNorm (one wave32 per node) ------------------------
__global__ void finalize_kernel(const float* __restrict__ h,
                                const float* __restrict__ gamma,
                                const float* __restrict__ beta,
                                float* __restrict__ out, int Nn) {
  const int lane = threadIdx.x & 31;
  const int node = blockIdx.x * (blockDim.x >> 5) + (threadIdx.x >> 5);
  if (node >= Nn) return;
  const size_t off = (size_t)node * FEAT;
  float x[FEAT / 32];
  float s1 = 0.0f, s2 = 0.0f;
#pragma unroll
  for (int i = 0; i < FEAT / 32; ++i) {
    float v = h[off + lane + 32 * i];
    v = (v > 0.0f) ? v : NEG_SLOPE * v;   // LeakyReLU(0.1)
    x[i] = v;
    s1 += v;
    s2 += v * v;
  }
#pragma unroll
  for (int d = 16; d > 0; d >>= 1) {      // wave32 butterfly reduction
    s1 += __shfl_xor(s1, d, 32);
    s2 += __shfl_xor(s2, d, 32);
  }
  const float mu  = s1 * (1.0f / FEAT);
  const float var = s2 * (1.0f / FEAT) - mu * mu;
  const float inv = rsqrtf(var + LN_EPS);
#pragma unroll
  for (int i = 0; i < FEAT / 32; ++i) {
    const int c = lane + 32 * i;
    out[off + c] = gamma[c] * (x[i] - mu) * inv + beta[c];
  }
}

// ---------------------------------------------------------------------------
static inline size_t align256(size_t x) { return (x + 255) & ~(size_t)255; }

extern "C" void kernel_launch(void* const* d_in, const int* in_sizes, int n_in,
                              void* d_out, int out_size, void* d_ws, size_t ws_size,
                              hipStream_t stream) {
  const float* node_feats = (const float*)d_in[0];
  const int*   src        = (const int*)d_in[1];
  const int*   dst        = (const int*)d_in[2];
  const int*   etype      = (const int*)d_in[3];
  const float* bases      = (const float*)d_in[4];
  const float* comp       = (const float*)d_in[5];
  const float* loopW      = (const float*)d_in[6];
  const float* bias       = (const float*)d_in[7];
  const float* gamma      = (const float*)d_in[8];
  const float* beta       = (const float*)d_in[9];
  float* out = (float*)d_out;

  const int Nn = in_sizes[0] / FEAT;                 // 50000 nodes
  const int E  = in_sizes[1];                        // 800000 edges
  const int Bn = in_sizes[4] / (FEAT * FEAT);        // 4 bases
  const int R  = in_sizes[5] / Bn;                   // 8 relations

  // Workspace layout (~130 MB total; T and h sized to stay L2-resident).
  char* ws = (char*)d_ws;
  size_t off = 0;
  unsigned short* Wt = (unsigned short*)(ws + off);
  off += align256((size_t)(R + 1) * FEAT * FEAT * sizeof(unsigned short));
  unsigned short* nodeBf = (unsigned short*)(ws + off);
  off += align256((size_t)Nn * FEAT * sizeof(unsigned short));
  float* T = (float*)(ws + off);
  off += align256((size_t)Nn * FEAT * sizeof(float));
  float* h = (float*)(ws + off);
  (void)ws_size;

  // 1) basis-mixed, transposed bf16 weights (+ self-loop weight at index R)
  {
    int total = (R + 1) * FEAT * FEAT;
    build_wt_kernel<<<(total + 255) / 256, 256, 0, stream>>>(
        bases, comp, loopW, Wt, R, Bn);
  }
  // 2) bf16 node features
  {
    int total = Nn * FEAT;
    to_bf16_kernel<<<(total + 255) / 256, 256, 0, stream>>>(
        node_feats, nodeBf, total);
  }

  const int gemmBlocks = (Nn + 63) / 64;

  // 3) self-loop transform: h = X @ loop_weight + bias
  gemm_bf16_kernel<<<gemmBlocks, 256, 0, stream>>>(
      nodeBf, Wt, /*widx=*/R, h, bias, /*addBias=*/1, Nn);

  // 4) stream relations: GEMM into T (L2-resident), then scatter into h
  for (int r = 0; r < R; ++r) {
    gemm_bf16_kernel<<<gemmBlocks, 256, 0, stream>>>(
        nodeBf, Wt, r, T, bias, /*addBias=*/0, Nn);
    scatter_rel_kernel<<<2048, 256, 0, stream>>>(T, h, src, dst, etype, E, r);
  }

  // 5) LeakyReLU + LayerNorm -> d_out
  finalize_kernel<<<(Nn + 7) / 8, 256, 0, stream>>>(h, gamma, beta, out, Nn);
  (void)n_in; (void)out_size;
}
